// FP8QDQLinear_62070867362248
// MI455X (gfx1250) — 8-bit matrix operand order repaired and checked by reading, compiled, not run on hardware
//
#include <hip/hip_runtime.h>
#include <cstdint>

typedef __attribute__((ext_vector_type(16))) int   v16i;
typedef __attribute__((ext_vector_type(8)))  float v8f;
typedef __attribute__((ext_vector_type(4)))  int   v4i;

#define TILE   128
#define K_TOT  4096
#define N_TOT  4096
#define NKT    (K_TOT / TILE)

#define AS1 __attribute__((address_space(1)))
#define AS3 __attribute__((address_space(3)))

#if defined(__has_builtin) && __has_builtin(__builtin_amdgcn_global_load_async_to_lds_b128)
#define HAVE_ASYNC_LDS 1
#else
#define HAVE_ASYNC_LDS 0
#endif

__device__ __forceinline__ int pack4_fp8(float x, float y, float z, float w) {
#if defined(__has_builtin) && __has_builtin(__builtin_amdgcn_cvt_pk_fp8_f32)
    int p = __builtin_amdgcn_cvt_pk_fp8_f32(x, y, 0, false);
    p = __builtin_amdgcn_cvt_pk_fp8_f32(z, w, p, true);
    return p;
#else
    int lo, hi;
    asm volatile("v_cvt_pk_fp8_f32 %0, %1, %2" : "=v"(lo) : "v"(x), "v"(y));
    asm volatile("v_cvt_pk_fp8_f32 %0, %1, %2" : "=v"(hi) : "v"(z), "v"(w));
    return (lo & 0xffff) | (hi << 16);
#endif
}

__device__ __forceinline__ float clamp_q(float v) {
    v *= 0.5f;
    return fminf(fmaxf(v, -448.0f), 448.0f);
}

__device__ __forceinline__ void async_copy16(const void* g, void* l) {
#if HAVE_ASYNC_LDS
    __builtin_amdgcn_global_load_async_to_lds_b128(
        (AS1 v4i*)(uintptr_t)g,
        (AS3 v4i*)(uint32_t)(uintptr_t)l, 0, 0);
#else
    *(int4*)l = *(const int4*)g;
#endif
}

__device__ __forceinline__ void wait_async_lds() {
#if HAVE_ASYNC_LDS
#if defined(__has_builtin) && __has_builtin(__builtin_amdgcn_s_wait_asynccnt)
    __builtin_amdgcn_s_wait_asynccnt(0);
#else
    asm volatile("s_wait_asynccnt 0x0" ::: "memory");
#endif
#endif
}

__global__ __launch_bounds__(256)
void quantize_fp8(const float* __restrict__ A, const float* __restrict__ W,
                  unsigned char* __restrict__ qa, unsigned char* __restrict__ qw,
                  int n4)
{
    const int stride = gridDim.x * blockDim.x;
    for (int i = blockIdx.x * blockDim.x + threadIdx.x; i < n4; i += stride) {
        float4 a = ((const float4*)A)[i];
        ((int*)qa)[i] = pack4_fp8(clamp_q(a.x), clamp_q(a.y),
                                  clamp_q(a.z), clamp_q(a.w));
        float4 w = ((const float4*)W)[i];
        ((int*)qw)[i] = pack4_fp8(w.x, w.y, w.z, w.w);
    }
}

__global__ __launch_bounds__(256)
void fp8_qdq_gemm(const unsigned char* __restrict__ Aq,
                  const unsigned char* __restrict__ Wq,
                  const float* __restrict__ B,
                  float* __restrict__ O)
{
    __shared__ __align__(16) unsigned char ldsA[2][TILE * TILE];
    __shared__ __align__(16) unsigned char ldsB[2][TILE * TILE];

    const int tid  = threadIdx.x;
    const int lane = tid & 31;
    const int wave = tid >> 5;
    const int l16  = lane & 15;
    const int lh   = lane >> 4;

    const int mBlk = blockIdx.y * TILE;
    const int nBlk = blockIdx.x * TILE;

    const int wm = wave & 1;
    const int wn = wave >> 1;

    v8f acc[4][2];
    #pragma unroll
    for (int i = 0; i < 4; ++i)
        #pragma unroll
        for (int j = 0; j < 2; ++j)
            #pragma unroll
            for (int e = 0; e < 8; ++e) acc[i][j][e] = 0.0f;

    const unsigned char* aBase = Aq + (size_t)mBlk * K_TOT;
    const unsigned char* wBase = Wq + (size_t)nBlk * K_TOT;

    auto stage = [&](int kt, int buf) {
        const unsigned char* aSrc = aBase + kt * TILE;
        const unsigned char* wSrc = wBase + kt * TILE;
        #pragma unroll
        for (int it = 0; it < 4; ++it) {
            int s   = tid + it * 256;
            int row = s >> 3;
            int c   = (s & 7) * 16;
            async_copy16(aSrc + (size_t)row * K_TOT + c, &ldsA[buf][row * TILE + c]);
            async_copy16(wSrc + (size_t)row * K_TOT + c, &ldsB[buf][row * TILE + c]);
        }
    };

    auto loadA = [&](int buf, int ms) -> v16i {
        v16i a;
        const unsigned char* base =
            &ldsA[buf][(wm * 64 + ms * 16 + l16) * TILE + 16 * lh];
        #pragma unroll
        for (int j = 0; j < 8; ++j) {
            long long v = *(const long long*)(base + 32 * (j >> 1) + 8 * (j & 1));
            a[2 * j]     = (int)(v & 0xffffffffLL);
            a[2 * j + 1] = (int)(v >> 32);
        }
        return a;
    };
    auto loadB = [&](int buf, int ns) -> v16i {
        v16i b;
        const unsigned char* base =
            &ldsB[buf][(wn * 32 + ns * 16 + l16) * TILE + 16 * lh];
        #pragma unroll
        for (int q = 0; q < 4; ++q) {
            int4 v = *(const int4*)(base + 32 * q);
            b[4 * q]     = v.x;
            b[4 * q + 1] = v.y;
            b[4 * q + 2] = v.z;
            b[4 * q + 3] = v.w;
        }
        return b;
    };

    stage(0, 0);
    wait_async_lds();
    __syncthreads();
    for (int kt = 0; kt < NKT; ++kt) {
        const int cur = kt & 1;
        if (kt + 1 < NKT) stage(kt + 1, cur ^ 1);

        v16i b0 = loadB(cur, 0);
        v16i b1 = loadB(cur, 1);
        #pragma unroll
        for (int ms = 0; ms < 4; ++ms) {
            v16i a = loadA(cur, ms);
            acc[ms][0] = __builtin_amdgcn_wmma_f32_16x16x128_fp8_fp8(
                a, b0, (short)0, acc[ms][0], false, false);
            acc[ms][1] = __builtin_amdgcn_wmma_f32_16x16x128_fp8_fp8(
                a, b1, (short)0, acc[ms][1], false, false);
        }
        wait_async_lds();
        __syncthreads();
    }

    const int col0 = nBlk + wn * 32 + l16;
    const float bv0 = B[col0];
    const float bv1 = B[col0 + 16];
    #pragma unroll
    for (int ms = 0; ms < 4; ++ms) {
        const int mrow = mBlk + wm * 64 + ms * 16 + 8 * lh;
        #pragma unroll
        for (int r = 0; r < 8; ++r) {
            O[(size_t)(mrow + r) * N_TOT + col0]      = acc[ms][0][r] * 4.0f + bv0;
            O[(size_t)(mrow + r) * N_TOT + col0 + 16] = acc[ms][1][r] * 4.0f + bv1;
        }
    }
}

extern "C" void kernel_launch(void* const* d_in, const int* in_sizes, int n_in,
                              void* d_out, int out_size, void* d_ws, size_t ws_size,
                              hipStream_t stream) {
    const float* inp  = (const float*)d_in[0];
    const float* wgt  = (const float*)d_in[1];
    const float* bias = (const float*)d_in[2];
    float* out = (float*)d_out;

    const int M  = in_sizes[0] / K_TOT;
    unsigned char* qa = (unsigned char*)d_ws;
    unsigned char* qw = qa + (size_t)M * K_TOT;

    const int n4 = (M * K_TOT) / 4;
    quantize_fp8<<<4096, 256, 0, stream>>>(inp, wgt, qa, qw, n4);

    dim3 grid(N_TOT / TILE, M / TILE);
    fp8_qdq_gemm<<<grid, 256, 0, stream>>>(qa, qw, bias, out);
}
